// MGCNLayerWrapper_7971459301982
// MI455X (gfx1250) — matrix operand order repaired and checked by reading, not compiled, not run on hardware
//
#include <hip/hip_runtime.h>
#include <hip/hip_bf16.h>
#include <math.h>

#define D       256
#define TM      32      // rows (edges or nodes) per block tile (2 M-tiles)
#define KSLAB   32      // K-step per WMMA (bf16 16x16x32)
#define WSTRIDE 48      // halves per WsT row (96B, 32B-aligned)

typedef __bf16 bf16_t;
typedef __attribute__((ext_vector_type(16))) __bf16 v16bf;
typedef __attribute__((ext_vector_type(4)))  __bf16 v4bf;
typedef __attribute__((ext_vector_type(8)))  float  v8f;

// Permutation that makes each lane's A-fragment contiguous in LDS.
// ISA A-layout (16x32 bf16, wave32): lane(m,g) half i -> k_local = (i<8 ? i : i+8) + 8g
// Store chunk order [0..7 | 16..23 | 8..15 | 24..31]: lane g reads halves [16g, 16g+16).
// 4-aligned runs stay contiguous, so float4 -> one packed v4bf (ds_store_b64).
static __device__ __forceinline__ constexpr int kpos(int j) {
  return (j < 8) ? j : (j < 16 ? j + 8 : (j < 24 ? j - 8 : j));
}

static __device__ __forceinline__ v4bf pack4(float x, float y, float z, float w) {
  v4bf q;
  q[0] = (bf16_t)x; q[1] = (bf16_t)y; q[2] = (bf16_t)z; q[3] = (bf16_t)w;
  return q;
}

// ---------------------------------------------------------------------------
// Small helper kernels
// ---------------------------------------------------------------------------
__global__ void zero_f32_kernel(float* __restrict__ p, int n) {
  const int i = blockIdx.x * blockDim.x + threadIdx.x;
  if (i < n) p[i] = 0.0f;
}

__global__ void degree_kernel(const int* __restrict__ edge_index,
                              float* __restrict__ deg, int E) {
  const int e = blockIdx.x * blockDim.x + threadIdx.x;
  if (e < E) unsafeAtomicAdd(&deg[edge_index[E + e]], 1.0f);  // dst row
}

__global__ void norm_kernel(float* __restrict__ deg, int n) {
  const int i = blockIdx.x * blockDim.x + threadIdx.x;
  if (i < n) deg[i] = 1.0f / fmaxf(deg[i], 1.0f);
}

// y = x + res * tanh(agg)
__global__ void finalize_kernel(const float* __restrict__ x,
                                const float* __restrict__ agg,
                                const float* __restrict__ res,
                                float* __restrict__ y, int total) {
  const int i = blockIdx.x * blockDim.x + threadIdx.x;
  if (i < total) y[i] = x[i] + res[0] * tanhf(agg[i]);
}

// ---------------------------------------------------------------------------
// W slab staging: WsT[col n][k_local] bf16 (fragment-contiguous for B frags).
// Thread owns a 4(k) x 8(n) patch -> 8x float4 global loads, 8x ds_store_b64.
// ---------------------------------------------------------------------------
static __device__ __forceinline__ void stage_w_slab(const float* __restrict__ W,
                                                    bf16_t* __restrict__ WsT,
                                                    int k0, int tid) {
  const int kr = (tid & 7) * 4;       // k_local group of 4
  const int cb = (tid >> 3) * 8;      // 8 contiguous cols
  const float4* r0 = (const float4*)(W + (size_t)(k0 + kr + 0) * D + cb);
  const float4* r1 = (const float4*)(W + (size_t)(k0 + kr + 1) * D + cb);
  const float4* r2 = (const float4*)(W + (size_t)(k0 + kr + 2) * D + cb);
  const float4* r3 = (const float4*)(W + (size_t)(k0 + kr + 3) * D + cb);
  const float4 a0 = r0[0], b0 = r0[1];
  const float4 a1 = r1[0], b1 = r1[1];
  const float4 a2 = r2[0], b2 = r2[1];
  const float4 a3 = r3[0], b3 = r3[1];
  bf16_t* base = &WsT[(size_t)cb * WSTRIDE + kpos(kr)];
  *(v4bf*)(base + 0 * WSTRIDE) = pack4(a0.x, a1.x, a2.x, a3.x);
  *(v4bf*)(base + 1 * WSTRIDE) = pack4(a0.y, a1.y, a2.y, a3.y);
  *(v4bf*)(base + 2 * WSTRIDE) = pack4(a0.z, a1.z, a2.z, a3.z);
  *(v4bf*)(base + 3 * WSTRIDE) = pack4(a0.w, a1.w, a2.w, a3.w);
  *(v4bf*)(base + 4 * WSTRIDE) = pack4(b0.x, b1.x, b2.x, b3.x);
  *(v4bf*)(base + 5 * WSTRIDE) = pack4(b0.y, b1.y, b2.y, b3.y);
  *(v4bf*)(base + 6 * WSTRIDE) = pack4(b0.z, b1.z, b2.z, b3.z);
  *(v4bf*)(base + 7 * WSTRIDE) = pack4(b0.w, b1.w, b2.w, b3.w);
}

static __device__ __forceinline__ v16bf ld_frag(const bf16_t* p) {
  return *(const v16bf*)p;  // 32B contiguous, 32B-aligned -> 2x ds_load_b128
}

// ---------------------------------------------------------------------------
// Dense GEMM with bias: out[M,256] = X[M,256] @ W[256,256] + bias
// Block: 256 threads (8 wave32), 32 rows; wave owns 32 cols x 32 rows
// -> 2 M-tiles x 2 N-tiles = 4 WMMA per k-step per wave.
// ---------------------------------------------------------------------------
__global__ __launch_bounds__(256) void gemm_bias_kernel(
    const float* __restrict__ X, const float* __restrict__ W,
    const float* __restrict__ bias, float* __restrict__ out, int M) {
  __shared__ bf16_t Af[8 * TM * 32];        // 16 KB
  __shared__ bf16_t WsT[D * WSTRIDE];       // 24 KB

  const int tid  = threadIdx.x;
  const int wave = tid >> 5, lane = tid & 31;
  const int rb   = blockIdx.x * TM;
  const int ncol = wave * 32;

  // Stage A: thread covers row r = tid>>3, k-chunk t = tid&7 (32 k's).
  {
    const int r = tid >> 3, t = tid & 7, c0 = t * 32;
    const int row = rb + r;
    bf16_t* dst = &Af[(t * TM + r) * 32];
    if (row < M) {
      const float4* p = (const float4*)(X + (size_t)row * D + c0);
#pragma unroll
      for (int j4 = 0; j4 < 8; ++j4) {
        const float4 v = p[j4];
        *(v4bf*)&dst[kpos(4 * j4)] = pack4(v.x, v.y, v.z, v.w);
      }
    } else {
#pragma unroll
      for (int j4 = 0; j4 < 8; ++j4) { v4bf z = {}; *(v4bf*)&dst[4 * j4] = z; }
    }
  }

  v8f acc[2][2] = {{{}, {}}, {{}, {}}};
  const int g = lane >> 4, ml = lane & 15;

  for (int k0 = 0; k0 < D; k0 += KSLAB) {
    __syncthreads();                      // WsT reuse (publishes Af on 1st pass)
    stage_w_slab(W, WsT, k0, tid);
    __syncthreads();

    const int t = k0 >> 5;
    const v16bf a0 = ld_frag(&Af[((t * TM) + 0  + ml) * 32 + g * 16]);
    const v16bf a1 = ld_frag(&Af[((t * TM) + 16 + ml) * 32 + g * 16]);
    const v16bf b0 = ld_frag(&WsT[(ncol + ml) * WSTRIDE + g * 16]);
    const v16bf b1 = ld_frag(&WsT[(ncol + 16 + ml) * WSTRIDE + g * 16]);
    acc[0][0] = __builtin_amdgcn_wmma_f32_16x16x32_bf16(false, a0, false, b0, (short)0, acc[0][0], false, false);
    acc[0][1] = __builtin_amdgcn_wmma_f32_16x16x32_bf16(false, a0, false, b1, (short)0, acc[0][1], false, false);
    acc[1][0] = __builtin_amdgcn_wmma_f32_16x16x32_bf16(false, a1, false, b0, (short)0, acc[1][0], false, false);
    acc[1][1] = __builtin_amdgcn_wmma_f32_16x16x32_bf16(false, a1, false, b1, (short)0, acc[1][1], false, false);
  }

  // Epilogue: C/D layout (n = lane&15, m = v + 8g); hoisted bias, fast path.
  const int c0 = ncol + ml;
  const float bias0 = bias[c0], bias1 = bias[c0 + 16];
  if (rb + TM <= M) {
#pragma unroll
    for (int mt = 0; mt < 2; ++mt) {
#pragma unroll
      for (int v = 0; v < 8; ++v) {
        const size_t row = rb + mt * 16 + v + g * 8;
        out[row * D + c0]      = acc[mt][0][v] + bias0;
        out[row * D + c0 + 16] = acc[mt][1][v] + bias1;
      }
    }
  } else {
#pragma unroll
    for (int mt = 0; mt < 2; ++mt) {
#pragma unroll
      for (int v = 0; v < 8; ++v) {
        const int row = rb + mt * 16 + v + g * 8;
        if (row < M) {
          out[(size_t)row * D + c0]      = acc[mt][0][v] + bias0;
          out[(size_t)row * D + c0 + 16] = acc[mt][1][v] + bias1;
        }
      }
    }
  }
}

// ---------------------------------------------------------------------------
// Edge message kernel: tile of 32 edges,
//   m = x[src] * rel[type]     (compose -> bf16 LDS, fragment-major)
//   t = m @ Wdir               (WMMA; Wdir = W_in for e<E/2 else W_out)
//   agg[dst] += t * norm[dst]  (f32 global atomics)
// ---------------------------------------------------------------------------
__global__ __launch_bounds__(256) void edge_msg_kernel(
    const float* __restrict__ X, const float* __restrict__ rel,
    const int* __restrict__ edge_index, const int* __restrict__ edge_type,
    const float* __restrict__ W_in, const float* __restrict__ W_out,
    const float* __restrict__ norm, float* __restrict__ agg, int E) {
  __shared__ bf16_t Af[8 * TM * 32];
  __shared__ bf16_t WsT[D * WSTRIDE];
  __shared__ int    dstL[TM];
  __shared__ float  nrmL[TM];

  const int tid  = threadIdx.x;
  const int wave = tid >> 5, lane = tid & 31;
  const int eb   = blockIdx.x * TM;
  const int halfE = E >> 1;
  const float* __restrict__ W = (eb < halfE) ? W_in : W_out;  // uniform per block
  const int ncol = wave * 32;

  // Stage composed messages: row r = tid>>3, k-chunk t = tid&7.
  {
    const int r = tid >> 3, t = tid & 7, c0 = t * 32;
    const int e = eb + r;
    bf16_t* dst = &Af[(t * TM + r) * 32];
    if (e < E) {
      const int s  = edge_index[e];
      const int dd = edge_index[E + e];
      const int ty = edge_type[e];
      if (t == 0) { dstL[r] = dd; nrmL[r] = norm[dd]; }
      const float4* px = (const float4*)(X   + (size_t)s  * D + c0);
      const float4* pr = (const float4*)(rel + (size_t)ty * D + c0);
#pragma unroll
      for (int j4 = 0; j4 < 8; ++j4) {
        const float4 vx = px[j4];
        const float4 vr = pr[j4];
        *(v4bf*)&dst[kpos(4 * j4)] =
            pack4(vx.x * vr.x, vx.y * vr.y, vx.z * vr.z, vx.w * vr.w);
      }
    } else {
      if (t == 0) { dstL[r] = -1; nrmL[r] = 0.0f; }
#pragma unroll
      for (int j4 = 0; j4 < 8; ++j4) { v4bf z = {}; *(v4bf*)&dst[4 * j4] = z; }
    }
  }

  v8f acc[2][2] = {{{}, {}}, {{}, {}}};
  const int g = lane >> 4, ml = lane & 15;

  for (int k0 = 0; k0 < D; k0 += KSLAB) {
    __syncthreads();
    stage_w_slab(W, WsT, k0, tid);
    __syncthreads();

    const int t = k0 >> 5;
    const v16bf a0 = ld_frag(&Af[((t * TM) + 0  + ml) * 32 + g * 16]);
    const v16bf a1 = ld_frag(&Af[((t * TM) + 16 + ml) * 32 + g * 16]);
    const v16bf b0 = ld_frag(&WsT[(ncol + ml) * WSTRIDE + g * 16]);
    const v16bf b1 = ld_frag(&WsT[(ncol + 16 + ml) * WSTRIDE + g * 16]);
    acc[0][0] = __builtin_amdgcn_wmma_f32_16x16x32_bf16(false, a0, false, b0, (short)0, acc[0][0], false, false);
    acc[0][1] = __builtin_amdgcn_wmma_f32_16x16x32_bf16(false, a0, false, b1, (short)0, acc[0][1], false, false);
    acc[1][0] = __builtin_amdgcn_wmma_f32_16x16x32_bf16(false, a1, false, b0, (short)0, acc[1][0], false, false);
    acc[1][1] = __builtin_amdgcn_wmma_f32_16x16x32_bf16(false, a1, false, b1, (short)0, acc[1][1], false, false);
  }

  // Scatter: norm-scale then atomic add into agg[dst]
  const int c0 = ncol + ml;
  if (eb + TM <= E) {
#pragma unroll
    for (int mt = 0; mt < 2; ++mt) {
#pragma unroll
      for (int v = 0; v < 8; ++v) {
        const int m = mt * 16 + v + g * 8;
        const int dd = dstL[m];
        const float nf = nrmL[m];
        unsafeAtomicAdd(&agg[(size_t)dd * D + c0],      acc[mt][0][v] * nf);
        unsafeAtomicAdd(&agg[(size_t)dd * D + c0 + 16], acc[mt][1][v] * nf);
      }
    }
  } else {
#pragma unroll
    for (int mt = 0; mt < 2; ++mt) {
#pragma unroll
      for (int v = 0; v < 8; ++v) {
        const int m = mt * 16 + v + g * 8;
        const int dd = dstL[m];
        if (dd >= 0) {
          const float nf = nrmL[m];
          unsafeAtomicAdd(&agg[(size_t)dd * D + c0],      acc[mt][0][v] * nf);
          unsafeAtomicAdd(&agg[(size_t)dd * D + c0 + 16], acc[mt][1][v] * nf);
        }
      }
    }
  }
}

// ---------------------------------------------------------------------------
// Host-side launch
// ---------------------------------------------------------------------------
extern "C" void kernel_launch(void* const* d_in, const int* in_sizes, int n_in,
                              void* d_out, int out_size, void* d_ws, size_t ws_size,
                              hipStream_t stream) {
  const float* emb        = (const float*)d_in[0];
  const int*   edge_index = (const int*)  d_in[1];
  const int*   edge_type  = (const int*)  d_in[2];
  const float* rel1    = (const float*)d_in[3];
  const float* W_in1   = (const float*)d_in[4];
  const float* W_out1  = (const float*)d_in[5];
  const float* W_loop1 = (const float*)d_in[6];
  const float* b1      = (const float*)d_in[7];
  const float* rel2    = (const float*)d_in[8];
  const float* W_in2   = (const float*)d_in[9];
  const float* W_out2  = (const float*)d_in[10];
  const float* W_loop2 = (const float*)d_in[11];
  const float* b2      = (const float*)d_in[12];
  const float* res     = (const float*)d_in[13];
  const float* mu_w    = (const float*)d_in[14];
  const float* mu_b    = (const float*)d_in[15];

  const int N = in_sizes[0] / D;   // nodes
  const int E = in_sizes[2];       // edges (edge_type length)
  float* out = (float*)d_out;

  // Workspace layout: deg/norm[N] | agg[N*D] | h1[N*D] | h2[N*D]
  char* ws = (char*)d_ws;
  size_t off = 0;
  auto align256 = [](size_t x) { return (x + 255) & ~(size_t)255; };
  float* deg = (float*)(ws + off); off += align256((size_t)N * 4);
  float* agg = (float*)(ws + off); off += align256((size_t)N * D * 4);
  float* h1  = (float*)(ws + off); off += align256((size_t)N * D * 4);
  float* h2  = (float*)(ws + off); off += align256((size_t)N * D * 4);
  (void)ws_size; (void)n_in; (void)out_size;

  const int total = N * D;
  const int gElem = (total + 255) / 256;
  const int gN    = (N + 255) / 256;
  const int gEth  = (E + 255) / 256;
  const int gM    = (N + TM - 1) / TM;   // GEMM row tiles
  const int gEb   = (E + TM - 1) / TM;   // edge tiles

  // Degrees -> in-degree norm (in place)
  zero_f32_kernel<<<gN, 256, 0, stream>>>(deg, N);
  degree_kernel<<<gEth, 256, 0, stream>>>(edge_index, deg, E);
  norm_kernel<<<gN, 256, 0, stream>>>(deg, N);

  // Layer 1: agg = emb @ W_loop1 + b1 ; += scatter(messages) ; h1 = emb + res*tanh(agg)
  gemm_bias_kernel<<<gM, 256, 0, stream>>>(emb, W_loop1, b1, agg, N);
  edge_msg_kernel<<<gEb, 256, 0, stream>>>(emb, rel1, edge_index, edge_type,
                                           W_in1, W_out1, deg, agg, E);
  finalize_kernel<<<gElem, 256, 0, stream>>>(emb, agg, res, h1, total);

  // Layer 2
  gemm_bias_kernel<<<gM, 256, 0, stream>>>(h1, W_loop2, b2, agg, N);
  edge_msg_kernel<<<gEb, 256, 0, stream>>>(h1, rel2, edge_index, edge_type,
                                           W_in2, W_out2, deg, agg, E);
  finalize_kernel<<<gElem, 256, 0, stream>>>(h1, agg, res, h2, total);

  // Drift head: out = h2 @ mu_w + mu_b
  gemm_bias_kernel<<<gM, 256, 0, stream>>>(h2, mu_w, mu_b, out, N);
}
